// ResnetBBlock_19713899888771
// MI455X (gfx1250) — matrix operand order repaired and checked by reading, not compiled, not run on hardware
//
#include <hip/hip_runtime.h>
#include <stdint.h>

// MI455X / gfx1250 KPConv ResNet bottleneck block.
// All contractions: v_wmma_f32_16x16x32_bf16 (bf16 operands, f32 accum).
// Weights pre-swizzled to fragment-ready bf16 buffers; blocks stage them into
// LDS via global_load_async_to_lds_b128 (ASYNCcnt) when the toolchain has it.
// Workspace requirement: ~96.3 MiB.

typedef __attribute__((ext_vector_type(16))) __bf16 v16bf;
typedef __attribute__((ext_vector_type(8)))  float  v8f;
typedef __attribute__((ext_vector_type(2)))  float  v2f;
typedef __attribute__((ext_vector_type(2)))  __bf16 v2bf;
typedef int v4i __attribute__((vector_size(16)));   // matches builtin param pointee

#define N_PTS   262144
#define NNEI    16
#define NKP     15
#define C_IN    128
#define D2      64
#define C_OUT   256
#define INV_PINF 25.0f     // 1 / (0.04 * 1.0)
#define SLOPE   0.2f
#define BN_EPS  1e-5f

#if defined(__has_builtin)
#if __has_builtin(__builtin_amdgcn_global_load_async_to_lds_b128)
#define HAVE_ASYNC_LDS 1
#endif
#endif
#ifndef HAVE_ASYNC_LDS
#define HAVE_ASYNC_LDS 0
#endif

// native f32 -> bf16 (RNE) conversions; backend emits v_cvt_*bf16* ops
__device__ __forceinline__ unsigned short f2bf(float f) {
  __bf16 b = (__bf16)f;
  unsigned short u; __builtin_memcpy(&u, &b, 2);
  return u;
}
__device__ __forceinline__ unsigned pack2bf(float a, float b) {
  v2f t; t[0] = a; t[1] = b;
  v2bf r = __builtin_convertvector(t, v2bf);
  unsigned u; __builtin_memcpy(&u, &r, 4);
  return u;
}
__device__ __forceinline__ float lrelu(float x) { return x >= 0.f ? x : SLOPE * x; }

__device__ __forceinline__ v8f wmma_bf16(v16bf a, v16bf b, v8f c) {
  return __builtin_amdgcn_wmma_f32_16x16x32_bf16(false, a, false, b, (short)0, c, false, false);
}

// 16B global -> LDS copy, async (ASYNCcnt path) when available
__device__ __forceinline__ void async_cp16(unsigned short* lds_dst, const unsigned short* gsrc) {
#if HAVE_ASYNC_LDS
  __builtin_amdgcn_global_load_async_to_lds_b128((v4i*)(void*)const_cast<unsigned short*>(gsrc),
                                                 (v4i*)(void*)lds_dst, 0, 0);
#else
  uint4 t; __builtin_memcpy(&t, gsrc, 16);
  __builtin_memcpy(lds_dst, &t, 16);
#endif
}
__device__ __forceinline__ void async_wait_all() {
#if HAVE_ASYNC_LDS
#if __has_builtin(__builtin_amdgcn_s_wait_asynccnt)
  __builtin_amdgcn_s_wait_asynccnt(0);
#else
  asm volatile("s_wait_asynccnt 0" ::: "memory");
#endif
#endif
}

// ---------------------------------------------------------------------------
// Generic weight pre-swizzle: row-major [K x (NT*16)] f32 -> bf16 B-fragments.
// Layout: ((kt*NT + nt)*32 + L)*16 + e ; row r = (e>>3)*16 + (L>>4)*8 + (e&7) -> k = kt*32 + r,
// col = nt*16 + (L&15).
__global__ __launch_bounds__(256) void k_prep_frag(const float* __restrict__ src,
                                                   unsigned short* __restrict__ dst,
                                                   int NT, int ncols) {
  int i = blockIdx.x * 256 + threadIdx.x;
  int e = i & 15, L = (i >> 4) & 31, f = i >> 9;
  int nt = f % NT, kt = f / NT;
  int k = kt * 32 + ((e >> 3) << 4) + ((L >> 4) << 3) + (e & 7);
  int n = nt * 16 + (L & 15);
  dst[i] = f2bf(src[k * ncols + n]);
}

// kp_weight [15][64][64] f32 -> bf16 B-frags, k padded to 16 with zeros.
// Layout: ((kk*4 + dt)*32 + L)*16 + e ; kk = k*2 + (c>=32).
__global__ __launch_bounds__(256) void k_prep_kpw(const float* __restrict__ kpw,
                                                  unsigned short* __restrict__ frag) {
  int t  = blockIdx.x * 256 + threadIdx.x;            // 65536 elements
  int e  = t & 15, L = (t >> 4) & 31, dt = (t >> 9) & 3, kk = (t >> 11) & 31;
  int k  = kk >> 1;
  int c  = (kk & 1) * 32 + ((e >> 3) << 4) + ((L >> 4) << 3) + (e & 7);
  int d  = dt * 16 + (L & 15);
  float v = (k < NKP) ? kpw[(k * D2 + c) * D2 + d] : 0.f;
  frag[t] = f2bf(v);
}

// ---------------------------------------------------------------------------
// GEMM1: t1[N,64] = x[N,128] @ W1[128,64]
__global__ __launch_bounds__(256) void k_gemm1(const float* __restrict__ x,
                                               const unsigned short* __restrict__ wfrag,
                                               float* __restrict__ t1) {
  __shared__ alignas(32) unsigned short bfrag[16 * 512];   // 16KB
  int tid = threadIdx.x;
  #pragma unroll
  for (int j = 0; j < 4; ++j) {
    int chunk = j * 256 + tid;                             // 1024 x 16B
    async_cp16(&bfrag[chunk * 8], wfrag + chunk * 8);
  }
  async_wait_all();
  __syncthreads();

  int wave = tid >> 5, lane = tid & 31;
  int ln = lane & 15, half = lane >> 4;
  int row0 = blockIdx.x * 128 + wave * 16;
  const float* xr = x + (long)(row0 + ln) * C_IN;

  v8f acc[4];
  #pragma unroll
  for (int nt = 0; nt < 4; ++nt)
    #pragma unroll
    for (int r = 0; r < 8; ++r) acc[nt][r] = 0.f;

  #pragma unroll
  for (int kt = 0; kt < 4; ++kt) {
    unsigned pk[8];
    #pragma unroll
    for (int v = 0; v < 8; ++v) {
      int k0 = kt * 32 + ((v >> 2) << 4) + half * 8 + ((v & 3) << 1);
      pk[v] = pack2bf(xr[k0], xr[k0 + 1]);
    }
    v16bf af; __builtin_memcpy(&af, pk, 32);
    #pragma unroll
    for (int nt = 0; nt < 4; ++nt) {
      v16bf bf; __builtin_memcpy(&bf, &bfrag[((kt * 4 + nt) * 32 + lane) * 16], 32);
      acc[nt] = wmma_bf16(af, bf, acc[nt]);
    }
  }
  #pragma unroll
  for (int nt = 0; nt < 4; ++nt)
    #pragma unroll
    for (int r = 0; r < 8; ++r)
      t1[(long)(row0 + r + half * 8) * D2 + nt * 16 + ln] = acc[nt][r];
}

// ---------------------------------------------------------------------------
// BatchNorm stats: one block per channel, deterministic LDS tree reduction.
// Emits fused affine: mr[c] = gamma*rstd, mr[nch+c] = beta - gamma*rstd*mean.
__global__ __launch_bounds__(256) void k_stats(const float* __restrict__ t, int nch,
                                               const float* __restrict__ gamma,
                                               const float* __restrict__ beta,
                                               float* __restrict__ mr) {
  __shared__ float ss[256], sq[256];
  int c = blockIdx.x, tid = threadIdx.x;
  float s = 0.f, q = 0.f;
  for (int n = tid; n < N_PTS; n += 256) {
    float v = t[(long)n * nch + c];
    s += v; q += v * v;
  }
  ss[tid] = s; sq[tid] = q; __syncthreads();
  for (int st = 128; st > 0; st >>= 1) {
    if (tid < st) { ss[tid] += ss[tid + st]; sq[tid] += sq[tid + st]; }
    __syncthreads();
  }
  if (tid == 0) {
    const float invn = 1.0f / (float)N_PTS;
    float mean = ss[0] * invn;
    float var  = sq[0] * invn - mean * mean;
    float A = gamma[c] * rsqrtf(var + BN_EPS);
    mr[c]       = A;
    mr[nch + c] = beta[c] - A * mean;
  }
}

// ---------------------------------------------------------------------------
// BN1 + LeakyReLU, f32 -> bf16 feature buffer h[N,64]
__global__ __launch_bounds__(256) void k_bnact1(const float* __restrict__ t1,
                                                const float* __restrict__ mr,
                                                unsigned short* __restrict__ h) {
  long i4 = ((long)blockIdx.x * 256 + threadIdx.x) * 4;
  float4 v = *reinterpret_cast<const float4*>(t1 + i4);
  int c0 = (int)(i4 & 63);
  float y0 = lrelu(fmaf(mr[c0 + 0], v.x, mr[64 + c0 + 0]));
  float y1 = lrelu(fmaf(mr[c0 + 1], v.y, mr[64 + c0 + 1]));
  float y2 = lrelu(fmaf(mr[c0 + 2], v.z, mr[64 + c0 + 2]));
  float y3 = lrelu(fmaf(mr[c0 + 3], v.w, mr[64 + c0 + 3]));
  unsigned long long pk = (unsigned long long)pack2bf(y0, y1)
                        | ((unsigned long long)pack2bf(y2, y3) << 32);
  __builtin_memcpy(h + i4, &pk, 8);
}

// ---------------------------------------------------------------------------
// KPConv core: one wave per 16-point tile.
// Stage 1 (per point): wf[k,c] = sum_a w[a,k]*h[idx[a],c] as 16x32x64 WMMA,
//   scattered to LDS in stage-2 A-fragment order.
// Stage 2: out[16,64] = WF[16,1024(k,c)] @ KPW[1024,64] -> 32 K-steps x 4 N-tiles.
__global__ __launch_bounds__(32) void k_kpconv(const float* __restrict__ pos,
                                               const int* __restrict__ idx,
                                               const float* __restrict__ kpoints,
                                               const unsigned short* __restrict__ h,
                                               const unsigned short* __restrict__ kpwfrag,
                                               float* __restrict__ out) {
  __shared__ alignas(32) unsigned short wf[32 * 512];   // 32KB
  __shared__ alignas(32) unsigned short nf[16 * 64];    // one point's 16x64 bf16 feats
  __shared__ float nd[256 * 3];                         // neighbor deltas

  int lane = threadIdx.x;
  int ln = lane & 15, half = lane >> 4;
  int base = blockIdx.x * 16;

  float kpx = 0.f, kpy = 0.f, kpz = 0.f;
  if (ln < NKP) { kpx = kpoints[ln * 3]; kpy = kpoints[ln * 3 + 1]; kpz = kpoints[ln * 3 + 2]; }

  // stage neighbor deltas; prefetch gathered feature rows for the whole tile
  #pragma unroll
  for (int i = 0; i < 8; ++i) {
    int pa = lane * 8 + i;
    int pt = pa >> 4, a = pa & 15;
    int g = base + pt;
    int j = idx[g * NNEI + a];
    float dx = 1e6f, dy = 1e6f, dz = 1e6f;
    if (j < N_PTS) {
      dx = pos[j * 3]     - pos[g * 3];
      dy = pos[j * 3 + 1] - pos[g * 3 + 1];
      dz = pos[j * 3 + 2] - pos[g * 3 + 2];
      __builtin_prefetch(h + (long)j * 64, 0, 3);       // global_prefetch_b8
    }
    nd[pa * 3] = dx; nd[pa * 3 + 1] = dy; nd[pa * 3 + 2] = dz;
  }
  __syncthreads();

  for (int pt = 0; pt < 16; ++pt) {
    // stage this point's neighbor features (zeros for shadow neighbors)
    {
      int a = lane >> 1, piece = lane & 1;
      int j = idx[(base + pt) * NNEI + a];
      if (j < N_PTS) {
        #pragma unroll
        for (int q = 0; q < 4; ++q)
          async_cp16(&nf[a * 64 + piece * 32 + q * 8], h + (long)j * 64 + piece * 32 + q * 8);
      } else {
        uint4 z = make_uint4(0u, 0u, 0u, 0u);
        #pragma unroll
        for (int q = 0; q < 4; ++q)
          __builtin_memcpy(&nf[a * 64 + piece * 32 + q * 8], &z, 16);
      }
    }
    async_wait_all();
    __syncthreads();

    // A-fragment: KP influence weights w[a, k=ln]; K elems 16..31 are pad zeros
    unsigned apk[8];
    #pragma unroll
    for (int v = 0; v < 4; ++v) {
      float wv[2];
      #pragma unroll
      for (int p = 0; p < 2; ++p) {
        int a = half * 8 + v * 2 + p;
        float dx = nd[(pt * 16 + a) * 3]     - kpx;
        float dy = nd[(pt * 16 + a) * 3 + 1] - kpy;
        float dz = nd[(pt * 16 + a) * 3 + 2] - kpz;
        float w = fmaxf(0.f, 1.f - sqrtf(dx * dx + dy * dy + dz * dz) * INV_PINF);
        wv[p] = (ln == 15) ? 0.f : w;
      }
      apk[v] = pack2bf(wv[0], wv[1]);
    }
    #pragma unroll
    for (int v = 4; v < 8; ++v) apk[v] = 0u;
    v16bf af; __builtin_memcpy(&af, apk, 32);

    #pragma unroll
    for (int ct = 0; ct < 4; ++ct) {
      unsigned bpk[8];
      #pragma unroll
      for (int v = 0; v < 8; ++v) {
        unsigned short u0 = 0, u1 = 0;
        if (v < 4) {
          u0 = nf[(half * 8 + v * 2)     * 64 + ct * 16 + ln];
          u1 = nf[(half * 8 + v * 2 + 1) * 64 + ct * 16 + ln];
        }
        bpk[v] = (unsigned)u0 | ((unsigned)u1 << 16);
      }
      v16bf bfr; __builtin_memcpy(&bfr, bpk, 32);
      v8f z = { 0.f, 0.f, 0.f, 0.f, 0.f, 0.f, 0.f, 0.f };
      v8f wacc = wmma_bf16(af, bfr, z);

      // scatter wf[k, c] into stage-2 A-fragment order
      #pragma unroll
      for (int r = 0; r < 8; ++r) {
        int k  = r + half * 8;
        int c  = ct * 16 + ln;
        int kk = k * 2 + (c >> 5);
        int cl = c & 31;
        int L  = pt + (((cl >> 3) & 1) << 4);
        int vv = ((cl >> 4) << 2) | ((cl >> 1) & 3);
        wf[kk * 512 + L * 16 + vv * 2 + (cl & 1)] = f2bf(wacc[r]);
      }
    }
    __syncthreads();
  }

  // stage 2: contract against kp_weight fragments (L2-resident)
  v8f oacc[4];
  #pragma unroll
  for (int dt = 0; dt < 4; ++dt)
    #pragma unroll
    for (int r = 0; r < 8; ++r) oacc[dt][r] = 0.f;

  for (int kk = 0; kk < 32; ++kk) {
    v16bf af2; __builtin_memcpy(&af2, &wf[kk * 512 + lane * 16], 32);
    #pragma unroll
    for (int dt = 0; dt < 4; ++dt) {
      v16bf bf2; __builtin_memcpy(&bf2, kpwfrag + ((kk * 4 + dt) * 32 + lane) * 16, 32);
      oacc[dt] = wmma_bf16(af2, bf2, oacc[dt]);
    }
  }
  #pragma unroll
  for (int dt = 0; dt < 4; ++dt)
    #pragma unroll
    for (int r = 0; r < 8; ++r)
      out[(long)(base + r + half * 8) * D2 + dt * 16 + ln] = oacc[dt][r];
}

// ---------------------------------------------------------------------------
// BN2+LeakyReLU fused into GEMM2: t2[N,256] = lrelu(bn(kpout)) @ W2[64,256]
__global__ __launch_bounds__(256) void k_gemm2(const float* __restrict__ kpout,
                                               const unsigned short* __restrict__ wfrag,
                                               const float* __restrict__ mr2,
                                               float* __restrict__ t2) {
  __shared__ alignas(32) unsigned short bfrag[32 * 512];  // 32KB
  __shared__ float pA[64], pB[64];
  int tid = threadIdx.x;
  if (tid < 64) { pA[tid] = mr2[tid]; pB[tid] = mr2[64 + tid]; }
  #pragma unroll
  for (int j = 0; j < 8; ++j) {
    int chunk = j * 256 + tid;                            // 2048 x 16B
    async_cp16(&bfrag[chunk * 8], wfrag + chunk * 8);
  }
  async_wait_all();
  __syncthreads();

  int wave = tid >> 5, lane = tid & 31;
  int ln = lane & 15, half = lane >> 4;
  int row0 = blockIdx.x * 128 + wave * 16;
  const float* kr = kpout + (long)(row0 + ln) * D2;

  v8f acc[16];
  #pragma unroll
  for (int nt = 0; nt < 16; ++nt)
    #pragma unroll
    for (int r = 0; r < 8; ++r) acc[nt][r] = 0.f;

  #pragma unroll
  for (int kt = 0; kt < 2; ++kt) {
    unsigned pk[8];
    #pragma unroll
    for (int v = 0; v < 8; ++v) {
      int k0 = kt * 32 + ((v >> 2) << 4) + half * 8 + ((v & 3) << 1);
      float y0 = lrelu(fmaf(pA[k0],     kr[k0],     pB[k0]));
      float y1 = lrelu(fmaf(pA[k0 + 1], kr[k0 + 1], pB[k0 + 1]));
      pk[v] = pack2bf(y0, y1);
    }
    v16bf af; __builtin_memcpy(&af, pk, 32);
    #pragma unroll
    for (int nt = 0; nt < 16; ++nt) {
      v16bf bf; __builtin_memcpy(&bf, &bfrag[((kt * 16 + nt) * 32 + lane) * 16], 32);
      acc[nt] = wmma_bf16(af, bf, acc[nt]);
    }
  }
  #pragma unroll
  for (int nt = 0; nt < 16; ++nt)
    #pragma unroll
    for (int r = 0; r < 8; ++r)
      t2[(long)(row0 + r + half * 8) * C_OUT + nt * 16 + ln] = acc[nt][r];
}

// ---------------------------------------------------------------------------
// Final: out = lrelu(bn3(t2)) + x @ Wsc   (t2 lives in d_out; updated in place)
__global__ __launch_bounds__(256) void k_final(const float* __restrict__ x,
                                               const unsigned short* __restrict__ wfrag,
                                               const float* __restrict__ mr3,
                                               float* __restrict__ out) {
  __shared__ alignas(32) unsigned short bfrag[32 * 512];  // one phase = 32KB
  __shared__ float pA[256], pB[256];
  int tid = threadIdx.x;
  pA[tid] = mr3[tid]; pB[tid] = mr3[256 + tid];

  int wave = tid >> 5, lane = tid & 31;
  int ln = lane & 15, half = lane >> 4;
  int row0 = blockIdx.x * 128 + wave * 16;
  const float* xr = x + (long)(row0 + ln) * C_IN;

  v8f acc[16];
  #pragma unroll
  for (int nt = 0; nt < 16; ++nt)
    #pragma unroll
    for (int r = 0; r < 8; ++r) acc[nt][r] = 0.f;

  for (int phase = 0; phase < 2; ++phase) {
    __syncthreads();
    #pragma unroll
    for (int j = 0; j < 8; ++j) {
      int chunk = j * 256 + tid;                          // 2048 x 16B
      async_cp16(&bfrag[chunk * 8], wfrag + (long)phase * 16384 + chunk * 8);
    }
    async_wait_all();
    __syncthreads();
    #pragma unroll
    for (int ktl = 0; ktl < 2; ++ktl) {
      int kt = phase * 2 + ktl;
      unsigned pk[8];
      #pragma unroll
      for (int v = 0; v < 8; ++v) {
        int k0 = kt * 32 + ((v >> 2) << 4) + half * 8 + ((v & 3) << 1);
        pk[v] = pack2bf(xr[k0], xr[k0 + 1]);
      }
      v16bf af; __builtin_memcpy(&af, pk, 32);
      #pragma unroll
      for (int nt = 0; nt < 16; ++nt) {
        v16bf bf; __builtin_memcpy(&bf, &bfrag[((ktl * 16 + nt) * 32 + lane) * 16], 32);
        acc[nt] = wmma_bf16(af, bf, acc[nt]);
      }
    }
  }
  #pragma unroll
  for (int nt = 0; nt < 16; ++nt)
    #pragma unroll
    for (int r = 0; r < 8; ++r) {
      int d = nt * 16 + ln;
      long o = (long)(row0 + r + half * 8) * C_OUT + d;
      float t2v = out[o];
      out[o] = lrelu(fmaf(pA[d], t2v, pB[d])) + acc[nt][r];
    }
}

// ---------------------------------------------------------------------------
extern "C" void kernel_launch(void* const* d_in, const int* in_sizes, int n_in,
                              void* d_out, int out_size, void* d_ws, size_t ws_size,
                              hipStream_t stream) {
  (void)in_sizes; (void)n_in; (void)out_size; (void)ws_size;
  const float* x    = (const float*)d_in[0];
  const float* pos  = (const float*)d_in[1];
  const int*   idx  = (const int*)d_in[2];
  const float* W1   = (const float*)d_in[3];
  const float* g1   = (const float*)d_in[4];
  const float* b1   = (const float*)d_in[5];
  const float* kpts = (const float*)d_in[6];
  const float* kpw  = (const float*)d_in[7];
  const float* gkp  = (const float*)d_in[8];
  const float* bkp  = (const float*)d_in[9];
  const float* W2   = (const float*)d_in[10];
  const float* g2   = (const float*)d_in[11];
  const float* b2   = (const float*)d_in[12];
  const float* Wsc  = (const float*)d_in[13];
  float* out = (float*)d_out;

  char* ws = (char*)d_ws;
  float*          t1      = (float*)(ws);                           // [N,64] f32; reused as kpconv out
  unsigned short* hbuf    = (unsigned short*)(ws + 67108864ull);    // [N,64] bf16
  unsigned short* kpwfrag = (unsigned short*)(ws + 100663296ull);   // 64Ki bf16
  unsigned short* w1frag  = (unsigned short*)(ws + 100794368ull);   // 8Ki bf16
  unsigned short* w2frag  = (unsigned short*)(ws + 100810752ull);   // 16Ki bf16
  unsigned short* wscfrag = (unsigned short*)(ws + 100843520ull);   // 32Ki bf16
  float*          mr1     = (float*)(ws + 100909056ull);            // 128 f32
  float*          mr2     = mr1 + 128;                              // 128 f32
  float*          mr3     = mr2 + 128;                              // 512 f32

  k_prep_kpw <<<256, 256, 0, stream>>>(kpw, kpwfrag);
  k_prep_frag<<<32,  256, 0, stream>>>(W1,  w1frag,  4,  D2);
  k_prep_frag<<<64,  256, 0, stream>>>(W2,  w2frag,  16, C_OUT);
  k_prep_frag<<<128, 256, 0, stream>>>(Wsc, wscfrag, 16, C_OUT);

  k_gemm1 <<<2048, 256, 0, stream>>>(x, w1frag, t1);
  k_stats <<<64, 256, 0, stream>>>(t1, D2, g1, b1, mr1);
  k_bnact1<<<16384, 256, 0, stream>>>(t1, mr1, hbuf);
  k_kpconv<<<16384, 32, 0, stream>>>(pos, idx, kpts, hbuf, kpwfrag, t1);
  k_stats <<<64, 256, 0, stream>>>(t1, D2, gkp, bkp, mr2);
  k_gemm2 <<<2048, 256, 0, stream>>>(t1, w2frag, mr2, out);
  k_stats <<<256, 256, 0, stream>>>(out, C_OUT, g2, b2, mr3);
  k_final <<<2048, 256, 0, stream>>>(x, wscfrag, mr3, out);
}
